// MLP_51745765982752
// MI455X (gfx1250) — compile-verified, matrix operand order checked by reading, not run on hardware
//
#include <hip/hip_runtime.h>
#include <hip/hip_bf16.h>

typedef __attribute__((ext_vector_type(16))) __bf16 v16bf;
typedef __attribute__((ext_vector_type(8)))  float  v8f;

#define DIM_IN  512
#define DIM_H   128
#define DIM_OUT 64

// ---------------------------------------------------------------------------
// Zero-fill (atomic accumulation targets must be zeroed every launch)
// ---------------------------------------------------------------------------
__global__ __launch_bounds__(256) void zero_fill(float4* __restrict__ p, int n4) {
    int i = blockIdx.x * 256 + threadIdx.x;
    if (i < n4) p[i] = make_float4(0.f, 0.f, 0.f, 0.f);
}

// ---------------------------------------------------------------------------
// GEMM1: h1 = x @ w1.T   [N,512] x [128,512]^T -> [N,128]
// bf16 WMMA, f32 accumulate. 256 threads = 8 waves; block covers 16 rows,
// wave w covers column tile [16w, 16w+16).
// A-operand layout (16-bit A 16x32): lane = 16*half + m; per lane two
// contiguous 8-elem chunks at k0=8*half and 16+8*half of row m.
// B[k][n] = w1[n][k], so row-major w1 feeds B with the same pattern (lane=n).
// ---------------------------------------------------------------------------
__global__ __launch_bounds__(256) void gemm1_bf16(const float* __restrict__ x,
                                                  const float* __restrict__ w1,
                                                  float* __restrict__ h1,
                                                  int n_nodes) {
    const int row0 = blockIdx.x * 16;
    const int wave = threadIdx.x >> 5;   // 0..7 -> column tile
    const int lane = threadIdx.x & 31;
    const int half = lane >> 4;          // 0/1
    const int l16  = lane & 15;
    const int col0 = wave * 16;

    const int rowA = min(row0 + l16, n_nodes - 1);          // clamp tail reads
    const float* __restrict__ arow = x  + (size_t)rowA         * DIM_IN;
    const float* __restrict__ brow = w1 + (size_t)(col0 + l16) * DIM_IN;

    v8f acc = {};
#pragma unroll 4
    for (int k = 0; k < DIM_IN; k += 32) {
        const int kA = k + half * 8;
        const int kB = k + 16 + half * 8;
        v16bf a, b;
#pragma unroll
        for (int j = 0; j < 8; ++j) {
            a[j]     = (__bf16)arow[kA + j];
            a[8 + j] = (__bf16)arow[kB + j];
            b[j]     = (__bf16)brow[kA + j];
            b[8 + j] = (__bf16)brow[kB + j];
        }
        acc = __builtin_amdgcn_wmma_f32_16x16x32_bf16(
            false, a, false, b, (short)0, acc, false, false);
    }

    // C/D layout: VGPR r -> row = row0 + r + 8*half, col = col0 + l16
#pragma unroll
    for (int r = 0; r < 8; ++r) {
        const int row = row0 + r + 8 * half;
        if (row < n_nodes)
            h1[(size_t)row * DIM_H + col0 + l16] = acc[r];
    }
}

// ---------------------------------------------------------------------------
// GEMM2: h2 = relu(h1_agg) @ w2.T   [N,128] x [64,128]^T -> [N,64]
// ReLU fused into the on-the-fly f32->bf16 conversion of the A operand.
// 128 threads = 4 waves; block covers 16 rows, wave w covers cols [16w,16w+16).
// ---------------------------------------------------------------------------
__global__ __launch_bounds__(128) void gemm2_bf16(const float* __restrict__ hagg,
                                                  const float* __restrict__ w2,
                                                  float* __restrict__ h2,
                                                  int n_nodes) {
    const int row0 = blockIdx.x * 16;
    const int wave = threadIdx.x >> 5;   // 0..3
    const int lane = threadIdx.x & 31;
    const int half = lane >> 4;
    const int l16  = lane & 15;
    const int col0 = wave * 16;

    const int rowA = min(row0 + l16, n_nodes - 1);
    const float* __restrict__ arow = hagg + (size_t)rowA         * DIM_H;
    const float* __restrict__ brow = w2   + (size_t)(col0 + l16) * DIM_H;

    v8f acc = {};
#pragma unroll
    for (int k = 0; k < DIM_H; k += 32) {
        const int kA = k + half * 8;
        const int kB = k + 16 + half * 8;
        v16bf a, b;
#pragma unroll
        for (int j = 0; j < 8; ++j) {
            a[j]     = (__bf16)fmaxf(arow[kA + j], 0.f);   // fused ReLU
            a[8 + j] = (__bf16)fmaxf(arow[kB + j], 0.f);
            b[j]     = (__bf16)brow[kA + j];
            b[8 + j] = (__bf16)brow[kB + j];
        }
        acc = __builtin_amdgcn_wmma_f32_16x16x32_bf16(
            false, a, false, b, (short)0, acc, false, false);
    }

#pragma unroll
    for (int r = 0; r < 8; ++r) {
        const int row = row0 + r + 8 * half;
        if (row < n_nodes)
            h2[(size_t)row * DIM_OUT + col0 + l16] = acc[r];
    }
}

// ---------------------------------------------------------------------------
// SpMM scatter: out[dst] += w * h[src]. One thread handles 4 features
// (float4 gather, 4x global_atomic_add_f32). Gathered rows are L2-resident
// (h1: 51 MB, h2: 26 MB << 192 MB L2).
// ---------------------------------------------------------------------------
template <int D>
__global__ __launch_bounds__(256) void spmm_scatter(const int*   __restrict__ esrc,
                                                    const int*   __restrict__ edst,
                                                    const float* __restrict__ ew,
                                                    const float* __restrict__ hin,
                                                    float*       __restrict__ hout,
                                                    int n_edges) {
    constexpr int CH = D / 4;                       // float4 chunks per edge
    const long long gid = (long long)blockIdx.x * 256 + threadIdx.x;
    const long long e   = gid / CH;
    const int       c   = (int)(gid % CH);
    if (e >= n_edges) return;

    const int   s = esrc[e];
    const int   d = edst[e];
    const float w = ew[e];

    const float4 v = *(const float4*)(hin + (size_t)s * D + c * 4);
    float* o = hout + (size_t)d * D + c * 4;
    unsafeAtomicAdd(o + 0, w * v.x);
    unsafeAtomicAdd(o + 1, w * v.y);
    unsafeAtomicAdd(o + 2, w * v.z);
    unsafeAtomicAdd(o + 3, w * v.w);
}

// ---------------------------------------------------------------------------
// Row-wise log_softmax over 64 cols: one wave32 per row, 2 elems/lane.
// ---------------------------------------------------------------------------
__global__ __launch_bounds__(256) void log_softmax64(const float* __restrict__ hin,
                                                     float* __restrict__ out,
                                                     int n_nodes) {
    const int row  = blockIdx.x * 8 + (threadIdx.x >> 5);
    const int lane = threadIdx.x & 31;
    if (row >= n_nodes) return;

    const float2 v = ((const float2*)(hin + (size_t)row * DIM_OUT))[lane];

    float m = fmaxf(v.x, v.y);
#pragma unroll
    for (int off = 16; off > 0; off >>= 1)
        m = fmaxf(m, __shfl_xor(m, off, 32));

    float s = __expf(v.x - m) + __expf(v.y - m);
#pragma unroll
    for (int off = 16; off > 0; off >>= 1)
        s += __shfl_xor(s, off, 32);

    const float ls = __logf(s);
    float2 o;
    o.x = v.x - m - ls;
    o.y = v.y - m - ls;
    ((float2*)(out + (size_t)row * DIM_OUT))[lane] = o;
}

// ---------------------------------------------------------------------------
// Orchestration
// ---------------------------------------------------------------------------
extern "C" void kernel_launch(void* const* d_in, const int* in_sizes, int n_in,
                              void* d_out, int out_size, void* d_ws, size_t ws_size,
                              hipStream_t stream) {
    const float* x    = (const float*)d_in[0];   // [N, 512]
    const float* w1   = (const float*)d_in[1];   // [128, 512]
    const float* w2   = (const float*)d_in[2];   // [64, 128]
    const int*   esrc = (const int*)  d_in[3];   // [E]
    const int*   edst = (const int*)  d_in[4];   // [E]
    const float* ew   = (const float*)d_in[5];   // [E]

    const int n_nodes = in_sizes[0] / DIM_IN;
    const int n_edges = in_sizes[3];

    // Workspace layout (floats): [h1_agg N*128][h2_agg N*64][h1_gemm N*128][h2_gemm N*64]
    float* h1_agg = (float*)d_ws;
    float* h2_agg = h1_agg + (size_t)n_nodes * DIM_H;
    float* h1_gmm = h2_agg + (size_t)n_nodes * DIM_OUT;
    float* h2_gmm = h1_gmm + (size_t)n_nodes * DIM_H;

    // 1) Zero both atomic accumulation buffers (contiguous N*192 floats)
    {
        const int n4 = n_nodes * (DIM_H + DIM_OUT) / 4;
        zero_fill<<<(n4 + 255) / 256, 256, 0, stream>>>((float4*)h1_agg, n4);
    }

    // 2) GEMM1: x @ w1.T -> h1_gmm
    gemm1_bf16<<<(n_nodes + 15) / 16, 256, 0, stream>>>(x, w1, h1_gmm, n_nodes);

    // 3) SpMM over 128 features: h1_gmm -> h1_agg
    {
        const long long total = (long long)n_edges * (DIM_H / 4);
        spmm_scatter<DIM_H><<<(int)((total + 255) / 256), 256, 0, stream>>>(
            esrc, edst, ew, h1_gmm, h1_agg, n_edges);
    }

    // 4) GEMM2 (ReLU fused): relu(h1_agg) @ w2.T -> h2_gmm
    gemm2_bf16<<<(n_nodes + 15) / 16, 128, 0, stream>>>(h1_agg, w2, h2_gmm, n_nodes);

    // 5) SpMM over 64 features: h2_gmm -> h2_agg
    {
        const long long total = (long long)n_edges * (DIM_OUT / 4);
        spmm_scatter<DIM_OUT><<<(int)((total + 255) / 256), 256, 0, stream>>>(
            esrc, edst, ew, h2_gmm, h2_agg, n_edges);
    }

    // 6) log_softmax -> d_out
    log_softmax64<<<(n_nodes + 7) / 8, 256, 0, stream>>>(h2_agg, (float*)d_out, n_nodes);
}
